// Decoder_39831526703225
// MI455X (gfx1250) — compile-verified, matrix operand order checked by reading, not run on hardware
//
#include <hip/hip_runtime.h>
#include <hip/hip_bf16.h>

// ---------------------------------------------------------------------------
// Fused ConvS2S-style decoder for MI455X (gfx1250, wave32, WMMA).
// One workgroup per batch; all 3 blocks fused; GEMMs on v_wmma_f32_16x16x32_f16.
// All f16 fragments are contiguous b128-pair loads (weights pre-transposed).
// ---------------------------------------------------------------------------

typedef _Float16 h16;
typedef __attribute__((ext_vector_type(16))) _Float16 v16h;
typedef __attribute__((ext_vector_type(8)))  _Float16 v8h;
typedef __attribute__((ext_vector_type(8)))  float    v8f;
typedef __attribute__((ext_vector_type(4)))  float    v4f;

#define B_N   16
#define T_N   64
#define HW_N  512
#define E_N   256
#define V_N   128
#define KW    3
#define NBLK  3
#define KC    (KW * E_N)      // 768, conv GEMM K

// LDS layout (bytes):
//   buf   : [64][512] f32  = 131072   (conv output, then scores/alpha, then logits)
//   a_pad : [66][256] f16  =  33792   (zero-padded activation for conv)
//   zbuf  : [64][256] f16  =  32768
//   hbuf  : [64][256] f16  =  32768
//   sbuf  : [64][256] f16  =  32768
#define SMEM_BYTES (131072 + 33792 + 32768 * 3)

// ---- WMMA fragment loaders -------------------------------------------------
// 16-bit A[16x32] / B[32x16] lane layout per CDNA5 ISA 7.12.2: for lane half
// hh, the 16 fragment elements are exactly halves [8*hh, 8*hh+8) and
// [16+8*hh, 16+8*hh+8) of the K=32 slice -> two contiguous 16-byte loads.
__device__ __forceinline__ v16h load_frag16(const h16* base, int hh) {
  v8h lo = *(const v8h*)(base + (hh << 3));
  v8h hi = *(const v8h*)(base + 16 + (hh << 3));
  v16h r;
#pragma unroll
  for (int j = 0; j < 8; ++j) { r[j] = lo[j]; r[j + 8] = hi[j]; }
  return r;
}

__device__ __forceinline__ v16h load_frag_f32(const float* base, int hh) {
  v4f a = *(const v4f*)(base + (hh << 3));
  v4f b = *(const v4f*)(base + (hh << 3) + 4);
  v4f c = *(const v4f*)(base + 16 + (hh << 3));
  v4f d = *(const v4f*)(base + 16 + (hh << 3) + 4);
  v16h r;
#pragma unroll
  for (int j = 0; j < 4; ++j) {
    r[j] = (h16)a[j]; r[j + 4] = (h16)b[j];
    r[j + 8] = (h16)c[j]; r[j + 12] = (h16)d[j];
  }
  return r;
}

#define WMMA_F16(acc, af, bf) \
  __builtin_amdgcn_wmma_f32_16x16x32_f16(false, (af), false, (bf), (short)0, (acc), false, false)

// ---- prep kernels ----------------------------------------------------------
// dec16 : [B][HW][E] f16  (scores B operand, contiguous in K=E)
// resT  : [B][E][HW] f16  (context B operand, contiguous in K=HW)
__global__ void __launch_bounds__(256) prep_data_kernel(
    const float* __restrict__ enc, const float* __restrict__ dec,
    h16* __restrict__ dec16, h16* __restrict__ resT, int n) {
  int i = blockIdx.x * 256 + threadIdx.x;
  if (i < n) {
    float d = dec[i];
    dec16[i] = (h16)d;
    int b = i / (HW_N * E_N);
    int r = i - b * (HW_N * E_N);
    int spos = r >> 8;            // r / E_N
    int e = r & (E_N - 1);
    resT[(size_t)b * HW_N * E_N + e * HW_N + spos] = (h16)(enc[i] + d);
  }
}

// cwT : [2E=512][KC=768] f16  (conv B operand, contiguous in K)
__global__ void __launch_bounds__(256) prep_w_kernel(
    const float* __restrict__ cw, const float* __restrict__ ww, const float* __restrict__ wo,
    h16* __restrict__ cwT, h16* __restrict__ ww16, h16* __restrict__ wo16) {
  int i = blockIdx.x * 256 + threadIdx.x;
  if (i < KC * 2 * E_N) {
    int n = i / KC, kk = i - n * KC;
    cwT[i] = (h16)cw[kk * (2 * E_N) + n];
  }
  if (i < E_N * E_N) ww16[i] = (h16)ww[i];
  if (i < V_N * E_N) wo16[i] = (h16)wo[i];
}

// ---- main fused decoder kernel --------------------------------------------
__global__ void __launch_bounds__(256) decoder_main_kernel(
    const h16* __restrict__ dec16, const h16* __restrict__ resT,
    const h16* __restrict__ cwT,   const h16* __restrict__ ww16,
    const h16* __restrict__ wo16,
    const float* __restrict__ conv_b, const float* __restrict__ W_b,
    const float* __restrict__ Wo_b,
    const float* __restrict__ embed, const int* __restrict__ labels,
    float* __restrict__ out) {
  extern __shared__ __align__(32) char smem[];
  float* buf  = (float*)smem;                                   // [64][512]
  h16* a_pad  = (h16*)(smem + 131072);                          // [66][256]
  h16* zbuf   = a_pad + 66 * E_N;                               // [64][256]
  h16* hbuf   = zbuf + T_N * E_N;                               // [64][256]
  h16* sbuf   = hbuf + T_N * E_N;                               // [64][256]

  const int b    = blockIdx.x;
  const int tid  = threadIdx.x;
  const int wave = tid >> 5;
  const int lane = tid & 31;
  const int lm   = lane & 15;   // fragment line (A row / B col / D col)
  const int hh   = lane >> 4;   // lane half

  const h16* decb = dec16 + (size_t)b * HW_N * E_N;
  const h16* resb = resT + (size_t)b * HW_N * E_N;

  // Warm L2/WGP$ with the shared weight streams (global_prefetch_b8).
  for (int i = tid * 64; i < KC * 2 * E_N; i += 256 * 64)
    __builtin_prefetch(cwT + i, 0, 3);
  for (int i = tid * 64; i < E_N * E_N; i += 256 * 64)
    __builtin_prefetch(ww16 + i, 0, 3);
  for (int i = tid * 64; i < V_N * E_N; i += 256 * 64)
    __builtin_prefetch(wo16 + i, 0, 3);

  // ---- init: s = embed[labels], a = s (with zero pad rows 0 and 65) --------
  for (int i = tid; i < T_N * E_N; i += 256) {
    int t = i >> 8, e = i & 255;
    h16 sv = (h16)embed[labels[b * T_N + t] * E_N + e];
    sbuf[i] = sv;
    a_pad[(t + 1) * E_N + e] = sv;
  }
  a_pad[tid] = (h16)0.f;                 // row 0  (tid covers 0..255)
  a_pad[65 * E_N + tid] = (h16)0.f;      // row 65
  __syncthreads();

  const int mt = wave & 3;               // M tile (4 tiles of 16 rows)
  const int nh = wave >> 2;              // N half (0 or 1)
  const int t0 = mt * 16 + lm;           // A-fragment source row (global t)

  for (int blk = 0; blk < NBLK; ++blk) {
    // ==== 1. conv as implicit-im2col GEMM: [64 x 768] x [768 x 512] ========
    for (int ntl = 0; ntl < 16; ++ntl) {
      const int nt = nh * 16 + ntl;
      const int n0 = nt * 16 + lm;
      v8f acc = {};
      for (int ks = 0; ks < 24; ++ks) {
        const int kk = ks * 32;
        const int kb = kk >> 8;          // conv tap (constant within K step)
        const int e0 = kk & 255;
        v16h af = load_frag16(&a_pad[(t0 + kb) * E_N + e0], hh);
        v16h bf = load_frag16(&cwT[(size_t)n0 * KC + kk], hh);
        acc = WMMA_F16(acc, af, bf);
      }
#pragma unroll
      for (int v = 0; v < 8; ++v)
        buf[(mt * 16 + v + (hh << 3)) * 512 + nt * 16 + lm] = acc[v];
    }
    __syncthreads();

    // ==== 2. GLU: z = (za+b) * sigmoid(zb+b) ================================
    for (int i = tid; i < T_N * E_N; i += 256) {
      int t = i >> 8, e = i & 255;
      float za = buf[t * 512 + e]       + conv_b[e];
      float zb = buf[t * 512 + 256 + e] + conv_b[256 + e];
      zbuf[i] = (h16)(za * __builtin_amdgcn_rcpf(1.f + __expf(-zb)));
    }
    __syncthreads();

    // ==== 3. h = z @ W^T + W_b + s : [64 x 256] x [256 x 256] ===============
    for (int ntl = 0; ntl < 8; ++ntl) {
      const int nt = nh * 8 + ntl;
      const int n0 = nt * 16 + lm;
      v8f acc = {};
      for (int ks = 0; ks < 8; ++ks) {
        v16h af = load_frag16(&zbuf[t0 * E_N + ks * 32], hh);
        v16h bf = load_frag16(&ww16[n0 * E_N + ks * 32], hh);
        acc = WMMA_F16(acc, af, bf);
      }
#pragma unroll
      for (int v = 0; v < 8; ++v) {
        int t = mt * 16 + v + (hh << 3);
        int n = nt * 16 + lm;
        hbuf[t * E_N + n] = (h16)(acc[v] + W_b[n] + (float)sbuf[t * E_N + n]);
      }
    }
    __syncthreads();

    // ==== 4. scores = h @ dec^T : [64 x 256] x [256 x 512] ==================
    for (int ntl = 0; ntl < 16; ++ntl) {
      const int nt = nh * 16 + ntl;
      const int n0 = nt * 16 + lm;     // spatial position
      v8f acc = {};
      for (int ks = 0; ks < 8; ++ks) {
        v16h af = load_frag16(&hbuf[t0 * E_N + ks * 32], hh);
        v16h bf = load_frag16(&decb[(size_t)n0 * E_N + ks * 32], hh);
        acc = WMMA_F16(acc, af, bf);
      }
#pragma unroll
      for (int v = 0; v < 8; ++v)
        buf[(mt * 16 + v + (hh << 3)) * 512 + nt * 16 + lm] = acc[v];
    }
    __syncthreads();

    // ==== 5. softmax over HW (512) per row, 4 lanes per row =================
    {
      int r = tid >> 2, q = tid & 3;
      float* row = buf + r * 512;
      float mx = -1e30f;
      for (int j = 0; j < 128; ++j) mx = fmaxf(mx, row[q * 128 + j]);
      mx = fmaxf(mx, __shfl_xor(mx, 1, 4));
      mx = fmaxf(mx, __shfl_xor(mx, 2, 4));
      float sum = 0.f;
      for (int j = 0; j < 128; ++j) {
        float e = __expf(row[q * 128 + j] - mx);
        row[q * 128 + j] = e;
        sum += e;
      }
      sum += __shfl_xor(sum, 1, 4);
      sum += __shfl_xor(sum, 2, 4);
      float inv = __builtin_amdgcn_rcpf(sum);
      for (int j = 0; j < 128; ++j) row[q * 128 + j] *= inv;
    }
    __syncthreads();

    // ==== 6. c = alpha @ residual, a_next = c + z : [64x512] x [512x256] ====
    for (int ntl = 0; ntl < 8; ++ntl) {
      const int nt = nh * 8 + ntl;
      const int n0 = nt * 16 + lm;     // embedding dim
      v8f acc = {};
      for (int ks = 0; ks < 16; ++ks) {
        v16h af = load_frag_f32(&buf[t0 * 512 + ks * 32], hh);
        v16h bf = load_frag16(&resb[(size_t)n0 * HW_N + ks * 32], hh);
        acc = WMMA_F16(acc, af, bf);
      }
#pragma unroll
      for (int v = 0; v < 8; ++v) {
        int t = mt * 16 + v + (hh << 3);
        int n = nt * 16 + lm;
        a_pad[(t + 1) * E_N + n] = (h16)(acc[v] + (float)zbuf[t * E_N + n]);
      }
    }
    __syncthreads();
  }

  // ==== 7. logits = a @ Wo^T + Wo_b : [64 x 256] x [256 x 128] ==============
  for (int ntl = 0; ntl < 4; ++ntl) {
    const int nt = nh * 4 + ntl;
    const int n0 = nt * 16 + lm;
    v8f acc = {};
    for (int ks = 0; ks < 8; ++ks) {
      v16h af = load_frag16(&a_pad[(t0 + 1) * E_N + ks * 32], hh);
      v16h bf = load_frag16(&wo16[n0 * E_N + ks * 32], hh);
      acc = WMMA_F16(acc, af, bf);
    }
#pragma unroll
    for (int v = 0; v < 8; ++v)
      buf[(mt * 16 + v + (hh << 3)) * 512 + nt * 16 + lm] = acc[v] + Wo_b[nt * 16 + lm];
  }
  __syncthreads();

  // ==== 8. log_softmax over V (128) per row, write out ======================
  {
    int r = tid >> 2, q = tid & 3;
    float* row = buf + r * 512;
    float mx = -1e30f;
    for (int j = 0; j < 32; ++j) mx = fmaxf(mx, row[q * 32 + j]);
    mx = fmaxf(mx, __shfl_xor(mx, 1, 4));
    mx = fmaxf(mx, __shfl_xor(mx, 2, 4));
    float sum = 0.f;
    for (int j = 0; j < 32; ++j) sum += __expf(row[q * 32 + j] - mx);
    sum += __shfl_xor(sum, 1, 4);
    sum += __shfl_xor(sum, 2, 4);
    float lse = mx + __logf(sum);
    for (int j = 0; j < 32; ++j)
      out[(size_t)b * T_N * V_N + r * V_N + q * 32 + j] = row[q * 32 + j] - lse;
  }
}

// ---------------------------------------------------------------------------
extern "C" void kernel_launch(void* const* d_in, const int* in_sizes, int n_in,
                              void* d_out, int out_size, void* d_ws, size_t ws_size,
                              hipStream_t stream) {
  const float* enc    = (const float*)d_in[0];
  const float* dec    = (const float*)d_in[1];
  const int*   labels = (const int*)d_in[2];
  const float* embed  = (const float*)d_in[3];
  const float* conv_w = (const float*)d_in[4];
  const float* conv_b = (const float*)d_in[5];
  const float* W_w    = (const float*)d_in[6];
  const float* W_b    = (const float*)d_in[7];
  const float* Wo_w   = (const float*)d_in[8];
  const float* Wo_b   = (const float*)d_in[9];
  float* out = (float*)d_out;

  // workspace carve-out (f16 staging, ~9.4 MB total)
  const int nBHWE = B_N * HW_N * E_N;          // 2,097,152
  h16* dec16 = (h16*)d_ws;
  h16* resT  = dec16 + nBHWE;
  h16* cwT   = resT + nBHWE;                   // 393,216
  h16* ww16  = cwT + KC * 2 * E_N;             //  65,536
  h16* wo16  = ww16 + E_N * E_N;               //  32,768

  hipFuncSetAttribute((const void*)decoder_main_kernel,
                      hipFuncAttributeMaxDynamicSharedMemorySize, SMEM_BYTES);

  prep_data_kernel<<<(nBHWE + 255) / 256, 256, 0, stream>>>(enc, dec, dec16, resT, nBHWE);
  prep_w_kernel<<<(KC * 2 * E_N + 255) / 256, 256, 0, stream>>>(
      conv_w, W_w, Wo_w, cwT, ww16, wo16);

  decoder_main_kernel<<<B_N, 256, SMEM_BYTES, stream>>>(
      dec16, resT, cwT, ww16, wo16, conv_b, W_b, Wo_b, embed, labels, out);
}
